// BeamSearchDecoder_5016521801830
// MI455X (gfx1250) — matrix operand order repaired and checked by reading, not compiled, not run on hardware
//
#include <hip/hip_runtime.h>
#include <hip/hip_bf16.h>
#include <math.h>

// ---------------- problem constants ----------------
#define CV    50257   // vocab
#define CE    512     // embed dim
#define CH    1024    // hidden
#define CS    128     // encoder length
#define CB    8       // beams
#define CT    64      // max length
#define SOS   1
#define EOS_T 2
#define NEGV  (-1e9f)
#define H3    3072    // 3*CH
#define KCAT  2048    // 2*CH
#define VPAD  50272   // 16*3142

typedef __bf16 bf16;
typedef __bf16 v16bf __attribute__((ext_vector_type(16)));
typedef __bf16 v8bf  __attribute__((ext_vector_type(8)));
typedef float  v8f   __attribute__((ext_vector_type(8)));

// ---------------- pre-pass: fp32 [K][N] -> bf16 transposed [Npad][K] ----------------
__global__ void k_transpose_cast(const float* __restrict__ in, bf16* __restrict__ out,
                                 int K, int N, int Npad) {
    __shared__ float tile[32][33];
    int tx = threadIdx.x, ty = threadIdx.y;
    int gn = blockIdx.x * 32 + tx;   // input col (output row)
    int gk = blockIdx.y * 32 + ty;   // input row (output col)
    tile[ty][tx] = (gn < N) ? in[(size_t)gk * N + gn] : 0.0f;
    __syncthreads();
    int orow = blockIdx.x * 32 + ty; // n
    int ocol = blockIdx.y * 32 + tx; // k
    if (orow < Npad)
        out[(size_t)orow * K + ocol] = (bf16)tile[tx][ty];
}

// ---------------- pre-pass: init beam state + zero padded activation rows ----------------
__global__ void k_init_state(const float* __restrict__ h0, float* h, int* tokens,
                             float* cum, int* len, int* fin, int* lat,
                             bf16* xbuf, bf16* hbuf, bf16* xcat) {
    int tid = threadIdx.x;
    for (int i = tid; i < CB * CT; i += 256) tokens[i] = 0;
    for (int i = tid; i < CB * CH; i += 256) h[i] = h0[i & (CH - 1)];
    if (tid < CB) { cum[tid] = 0.0f; len[tid] = 1; fin[tid] = 0; lat[tid] = SOS; }
    for (int i = tid; i < 16 * CE;   i += 256) xbuf[i] = (bf16)0.0f;
    for (int i = tid; i < 16 * CH;   i += 256) hbuf[i] = (bf16)0.0f;
    for (int i = tid; i < 16 * KCAT; i += 256) xcat[i] = (bf16)0.0f;
}

// ---------------- per step: gather embedding + cast h to bf16 ----------------
__global__ void k_gather_cast(const float* __restrict__ emb, const float* __restrict__ h,
                              const int* __restrict__ lat,
                              bf16* __restrict__ xbuf, bf16* __restrict__ hbuf) {
    int id = blockIdx.x * 256 + threadIdx.x;     // 8192 threads
    int b = id >> 10, j = id & (CH - 1);
    hbuf[b * CH + j] = (bf16)h[b * CH + j];
    if (j < CE) {
        int tok = lat[b];
        xbuf[b * CE + j] = (bf16)emb[(size_t)tok * CE + j];
    }
}

// ---------------- WMMA GEMM: C[16][N] = A[16][K](bf16) @ Bt[N][K]^T(bf16) ----------------
// One wave per 16-column tile; A and B both use contiguous k = 16*half + s order.
template <int K>
__global__ void k_wmma_gemm(const bf16* __restrict__ A, const bf16* __restrict__ Bt,
                            float* __restrict__ C, int N) {
    int wave = threadIdx.x >> 5;
    int lane = threadIdx.x & 31;
    int tile = blockIdx.x * 8 + wave;
    int ntiles = N >> 4;
    if (tile >= ntiles) return;
    int n0   = tile << 4;
    int half = lane >> 4;     // 0 or 1
    int sub  = lane & 15;     // M for A, N-col for B

    const bf16* ap = A + (size_t)sub * K + half * 16;          // A: k0+half*16 .. +15
    const bf16* bp = Bt + (size_t)(n0 + sub) * K + half * 16;  // B: k0+half*16 .. +15

    v8f acc = {};
    for (int k0 = 0; k0 < K; k0 += 32) {
        __builtin_prefetch(bp + k0 + 256, 0, 1);
        v16bf a = *(const v16bf*)(ap + k0);
        v16bf b = *(const v16bf*)(bp + k0);
        acc = __builtin_amdgcn_wmma_f32_16x16x32_bf16(
                  false, a, false, b, (short)0, acc, false, false);
    }
    // D layout: VGPR i -> (M = half*8 + i, Ncol = n0 + sub)
    #pragma unroll
    for (int i = 0; i < 8; ++i)
        C[(size_t)(half * 8 + i) * N + n0 + sub] = acc[i];
}

// ---------------- per step: GRU gate math -> h_new (fp32) + xcat[:, :H] (bf16) --------
__global__ void k_gru(const float* __restrict__ gx, const float* __restrict__ gh,
                      const float* __restrict__ h, float* __restrict__ h_new,
                      bf16* __restrict__ xcat) {
    int id = blockIdx.x * 256 + threadIdx.x;  // 8192
    int b = id >> 10, j = id & (CH - 1);
    const float* gxr = gx + (size_t)b * H3;
    const float* ghr = gh + (size_t)b * H3;
    float xz = gxr[j], xr = gxr[CH + j], xn = gxr[2 * CH + j];
    float hz = ghr[j], hr = ghr[CH + j], hn = ghr[2 * CH + j];
    float z = 1.0f / (1.0f + expf(-(xz + hz)));
    float r = 1.0f / (1.0f + expf(-(xr + hr)));
    float n = tanhf(xn + r * hn);
    float hv = (1.0f - z) * n + z * h[b * CH + j];
    h_new[b * CH + j] = hv;
    xcat[b * KCAT + j] = (bf16)hv;
}

// ---------------- per step: attention softmax + context -> xcat[:, H:2H] --------------
__global__ void k_attention(const float* __restrict__ h_new, const float* __restrict__ enc,
                            bf16* __restrict__ xcat) {
    int b = blockIdx.x, s = threadIdx.x;  // 128 threads
    __shared__ float sc[CS];
    __shared__ float red[CS];
    const float* hb = h_new + (size_t)b * CH;
    const float* es = enc + (size_t)s * CH;
    float d = 0.0f;
    for (int k = 0; k < CH; ++k) d += hb[k] * es[k];
    sc[s] = d; red[s] = d; __syncthreads();
    for (int o = 64; o > 0; o >>= 1) { if (s < o) red[s] = fmaxf(red[s], red[s + o]); __syncthreads(); }
    float m = red[0]; __syncthreads();
    float e = expf(sc[s] - m);
    sc[s] = e; red[s] = e; __syncthreads();
    for (int o = 64; o > 0; o >>= 1) { if (s < o) red[s] += red[s + o]; __syncthreads(); }
    float inv = 1.0f / red[0]; __syncthreads();
    sc[s] *= inv; __syncthreads();
    for (int j = s; j < CH; j += CS) {
        float c = 0.0f;
        for (int t = 0; t < CS; ++t) c += sc[t] * enc[(size_t)t * CH + j];
        xcat[b * KCAT + CH + j] = (bf16)c;
    }
}

// ---------------- per step: log-softmax + per-beam top-8 --------------------------------
__global__ void k_topk(const float* __restrict__ logits, const float* __restrict__ bias,
                       float* __restrict__ tklp, int* __restrict__ tkid) {
    int b = blockIdx.x, tid = threadIdx.x;  // 256 threads
    __shared__ float rv[256];
    __shared__ int   ri[256];
    __shared__ float s_m, s_lse;
    __shared__ int   chos[8];
    const float* L = logits + (size_t)b * VPAD;

    float lm = -3.4e38f;
    for (int j = tid; j < CV; j += 256) lm = fmaxf(lm, L[j] + bias[j]);
    rv[tid] = lm; __syncthreads();
    for (int o = 128; o > 0; o >>= 1) { if (tid < o) rv[tid] = fmaxf(rv[tid], rv[tid + o]); __syncthreads(); }
    if (tid == 0) s_m = rv[0];
    __syncthreads();
    float m = s_m, ls = 0.0f;
    for (int j = tid; j < CV; j += 256) ls += expf(L[j] + bias[j] - m);
    rv[tid] = ls; __syncthreads();
    for (int o = 128; o > 0; o >>= 1) { if (tid < o) rv[tid] += rv[tid + o]; __syncthreads(); }
    if (tid == 0) s_lse = m + logf(rv[0]);
    __syncthreads();
    float lse = s_lse;

    for (int r = 0; r < CB; ++r) {
        float bv = -3.4e38f; int bi = 0x7fffffff;
        for (int j = tid; j < CV; j += 256) {
            bool skip = false;
            for (int q = 0; q < r; ++q) skip |= (chos[q] == j);
            if (skip) continue;
            float v = L[j] + bias[j];
            if (v > bv || (v == bv && j < bi)) { bv = v; bi = j; }
        }
        rv[tid] = bv; ri[tid] = bi; __syncthreads();
        for (int o = 128; o > 0; o >>= 1) {
            if (tid < o) {
                if (rv[tid + o] > rv[tid] || (rv[tid + o] == rv[tid] && ri[tid + o] < ri[tid])) {
                    rv[tid] = rv[tid + o]; ri[tid] = ri[tid + o];
                }
            }
            __syncthreads();
        }
        if (tid == 0) { chos[r] = ri[0]; tklp[b * CB + r] = rv[0] - lse; tkid[b * CB + r] = ri[0]; }
        __syncthreads();
    }
}

// ---------------- per step: beam bookkeeping + state permutation ------------------------
__global__ void k_beam_update(int step, const float* __restrict__ tklp,
                              const int* __restrict__ tkid, int* tokens, float* cum,
                              int* len, int* fin, int* lat,
                              const float* __restrict__ h_new, float* __restrict__ h) {
    int tid = threadIdx.x;  // 256 threads
    __shared__ float s_cum[CB];
    __shared__ int   s_len[CB], s_fin[CB], s_lat[CB];
    __shared__ int   s_tok_old[CB * CT];
    __shared__ int   s_parent[CB], s_newtok[CB], s_fin_n[CB], s_len_n[CB], s_lat_n[CB];
    __shared__ float s_cum_n[CB];

    if (tid < CB) { s_cum[tid] = cum[tid]; s_len[tid] = len[tid]; s_fin[tid] = fin[tid]; s_lat[tid] = lat[tid]; }
    for (int i = tid; i < CB * CT; i += 256) s_tok_old[i] = tokens[i];
    __syncthreads();

    if (tid == 0) {
        float cc[CB * CB]; int cl[CB * CB]; float sco[CB * CB];
        for (int p = 0; p < CB; ++p) {
            bool alive = (s_fin[p] == 0);
            for (int c = 0; c < CB; ++c) {
                int i = p * CB + c;
                cc[i] = s_cum[p] + (alive ? tklp[i] : 0.0f);
                cl[i] = s_len[p] + (alive ? 1 : 0);
                float scr = cc[i] / (float)cl[i];
                bool valid = (alive || c == 0) && (step > 0 || p == 0);
                sco[i] = valid ? scr : NEGV;
            }
        }
        bool used[CB * CB];
        for (int i = 0; i < CB * CB; ++i) used[i] = false;
        for (int r = 0; r < CB; ++r) {
            int bi = 0; float bv = -3.5e38f;
            for (int i = 0; i < CB * CB; ++i)
                if (!used[i] && sco[i] > bv) { bv = sco[i]; bi = i; }
            used[bi] = true;
            int p = bi / CB, c = bi % CB;
            int ntok = tkid[p * CB + c];
            int pfin = s_fin[p];
            s_parent[r] = p;
            s_newtok[r] = ntok;
            s_cum_n[r] = cc[bi];
            s_len_n[r] = cl[bi];
            s_fin_n[r] = (pfin || ntok == EOS_T) ? 1 : 0;
            s_lat_n[r] = pfin ? s_lat[p] : ntok;
            (void)c;
        }
    }
    __syncthreads();

    // tokens_n[b][t] = (t == p_len-1 && !p_fin) ? new_tok : tokens_old[parent][t]
    for (int i = tid; i < CB * CT; i += 256) {
        int b = i >> 6, t = i & (CT - 1);
        int p = s_parent[b];
        int pl = s_len[p], pf = s_fin[p];
        tokens[i] = (t == pl - 1 && !pf) ? s_newtok[b] : s_tok_old[p * CT + t];
    }
    if (tid < CB) {
        cum[tid] = s_cum_n[tid]; len[tid] = s_len_n[tid];
        fin[tid] = s_fin_n[tid]; lat[tid] = s_lat_n[tid];
    }
    // h[b] = h_new[parent[b]]
    for (int i = tid; i < CB * CH; i += 256) {
        int b = i >> 10, j = i & (CH - 1);
        h[i] = h_new[(size_t)s_parent[b] * CH + j];
    }
}

// ---------------- final: pick best beam, emit tokens + score ----------------------------
__global__ void k_finalize(const int* __restrict__ tokens, const float* __restrict__ cum,
                           const int* __restrict__ len, const int* __restrict__ fin,
                           float* __restrict__ out) {
    __shared__ int s_best;
    __shared__ float s_score;
    if (threadIdx.x == 0) {
        bool any = false;
        for (int b = 0; b < CB; ++b) any |= (fin[b] != 0);
        float bv = -3.5e38f; int bi = 0;
        for (int b = 0; b < CB; ++b) {
            float sc = cum[b] / (float)len[b];
            float adj = any ? (fin[b] ? sc : NEGV) : sc;
            if (adj > bv) { bv = adj; bi = b; }
        }
        s_best = bi;
        s_score = cum[bi] / (float)len[bi];
    }
    __syncthreads();
    int t = threadIdx.x;
    if (t < CT) out[t] = (float)tokens[s_best * CT + t];
    if (t == 0) out[CT] = s_score;
}

// ---------------- host side ----------------
extern "C" void kernel_launch(void* const* d_in, const int* in_sizes, int n_in,
                              void* d_out, int out_size, void* d_ws, size_t ws_size,
                              hipStream_t stream) {
    const float* emb  = (const float*)d_in[0];
    const float* Wx   = (const float*)d_in[1];
    const float* Wh   = (const float*)d_in[2];
    const float* Wout = (const float*)d_in[3];
    const float* bout = (const float*)d_in[4];
    const float* enc  = (const float*)d_in[5];
    const float* h0   = (const float*)d_in[6];

    char* base = (char*)d_ws;
    size_t off = 0;
    auto carve = [&](size_t bytes) -> void* {
        void* p = base + off;
        off = (off + bytes + 255) & ~(size_t)255;
        return p;
    };
    bf16*  Wxt    = (bf16*)carve((size_t)H3 * CE * 2);
    bf16*  Wht    = (bf16*)carve((size_t)H3 * CH * 2);
    bf16*  WoutT  = (bf16*)carve((size_t)VPAD * KCAT * 2);
    bf16*  xbuf   = (bf16*)carve(16 * CE * 2);
    bf16*  hbuf   = (bf16*)carve(16 * CH * 2);
    bf16*  xcat   = (bf16*)carve(16 * KCAT * 2);
    float* gx     = (float*)carve((size_t)16 * H3 * 4);
    float* gh     = (float*)carve((size_t)16 * H3 * 4);
    float* h      = (float*)carve(CB * CH * 4);
    float* h_new  = (float*)carve(CB * CH * 4);
    float* logits = (float*)carve((size_t)16 * VPAD * 4);
    float* tklp   = (float*)carve(CB * CB * 4);
    int*   tkid   = (int*)carve(CB * CB * 4);
    int*   tokens = (int*)carve(CB * CT * 4);
    float* cum    = (float*)carve(CB * 4);
    int*   len    = (int*)carve(CB * 4);
    int*   fin    = (int*)carve(CB * 4);
    int*   lat    = (int*)carve(CB * 4);

    dim3 tb(32, 32);
    k_transpose_cast<<<dim3(H3 / 32, CE / 32), tb, 0, stream>>>(Wx, Wxt, CE, H3, H3);
    k_transpose_cast<<<dim3(H3 / 32, CH / 32), tb, 0, stream>>>(Wh, Wht, CH, H3, H3);
    k_transpose_cast<<<dim3(VPAD / 32, KCAT / 32), tb, 0, stream>>>(Wout, WoutT, KCAT, CV, VPAD);
    k_init_state<<<1, 256, 0, stream>>>(h0, h, tokens, cum, len, fin, lat, xbuf, hbuf, xcat);

    const int gates_blocks  = (H3 / 16 + 7) / 8;    // 24
    const int logits_blocks = (VPAD / 16 + 7) / 8;  // 393

    for (int t = 0; t < CT; ++t) {
        k_gather_cast<<<32, 256, 0, stream>>>(emb, h, lat, xbuf, hbuf);
        k_wmma_gemm<CE><<<gates_blocks, 256, 0, stream>>>(xbuf, Wxt, gx, H3);
        k_wmma_gemm<CH><<<gates_blocks, 256, 0, stream>>>(hbuf, Wht, gh, H3);
        k_gru<<<32, 256, 0, stream>>>(gx, gh, h, h_new, xcat);
        k_attention<<<CB, CS, 0, stream>>>(h_new, enc, xcat);
        k_wmma_gemm<KCAT><<<logits_blocks, 256, 0, stream>>>(xcat, WoutT, logits, VPAD);
        k_topk<<<CB, 256, 0, stream>>>(logits, bout, tklp, tkid);
        k_beam_update<<<1, 256, 0, stream>>>(t, tklp, tkid, tokens, cum, len, fin, lat, h_new, h);
    }
    k_finalize<<<1, 64, 0, stream>>>(tokens, cum, len, fin, (float*)d_out);
    (void)in_sizes; (void)n_in; (void)out_size; (void)ws_size;
}
